// RSNN_d_71760313582242
// MI455X (gfx1250) — matrix operand order repaired and checked by reading, not compiled, not run on hardware
//
#include <hip/hip_runtime.h>

#define NUM_INPUT 700
#define KP        704              // K padded to multiple of 32
#define H         256
#define NOUT      20
#define WIN       50
#define BATCH     512
#define NLAYERS   2
#define THRESH    0.3f
#define ALPHA     0.8f

typedef __attribute__((ext_vector_type(16))) __bf16 v16bf;
typedef __attribute__((ext_vector_type(8)))  __bf16 v8bf;
typedef __attribute__((ext_vector_type(8)))  float  v8f;
typedef __attribute__((ext_vector_type(4)))  unsigned int u32x4;
typedef __attribute__((ext_vector_type(8)))  int    i32x8;
typedef __attribute__((ext_vector_type(4)))  int    i32x4;

#if __has_builtin(__builtin_amdgcn_tensor_load_to_lds) && __has_builtin(__builtin_amdgcn_s_wait_tensorcnt)
#define USE_TDM 1
#else
#define USE_TDM 0
#endif

static __device__ __forceinline__ __bf16 bf_zero() {
    return __builtin_bit_cast(__bf16, (unsigned short)0);
}
static __device__ __forceinline__ __bf16 f2bf(float f) {
    unsigned u = __builtin_bit_cast(unsigned, f);
    unsigned r = u + 0x7FFFu + ((u >> 16) & 1u);   // round-to-nearest-even
    return __builtin_bit_cast(__bf16, (unsigned short)(r >> 16));
}
static __device__ __forceinline__ float bf2f(__bf16 b) { return (float)b; }

static __device__ __forceinline__ v8f wmma_bf16(v16bf a, v16bf b, v8f c) {
    // D = A(16x32 bf16) * B(32x16 bf16) + C(16x16 f32)
    return __builtin_amdgcn_wmma_f32_16x16x32_bf16(
        false, a, false, b, (short)0, c, false, false);
}

// A fragment 16x32 bf16 from row-major bf16 (base pre-offset to row0,k0).
// Lane m=lane&15 holds row m: elems {kh..kh+7} then {kh+16..kh+23}, kh=(lane>>4)*8.
static __device__ __forceinline__ v16bf load_a_bf16(const __bf16* base, int ld, int lane) {
    int m = lane & 15, kh = (lane >> 4) * 8;
    const __bf16* p = base + (size_t)m * ld + kh;
    v8bf lo = *(const v8bf*)(p);
    v8bf hi = *(const v8bf*)(p + 16);
    v16bf r;
#pragma unroll
    for (int i = 0; i < 8; ++i) { r[i] = lo[i]; r[8 + i] = hi[i]; }
    return r;
}

// B fragment 32x16 bf16 from row-major W[n][k] (base pre-offset to n0,k0).
static __device__ __forceinline__ v16bf load_b_bf16(const __bf16* base, int ld, int lane) {
    int n = lane & 15, ks = (lane >> 4) * 8;
    const __bf16* p = base + (size_t)n * ld + ks;
    v8bf lo = *(const v8bf*)(p);
    v8bf hi = *(const v8bf*)(p + 16);
    v16bf r;
#pragma unroll
    for (int i = 0; i < 8; ++i) { r[i] = lo[i]; r[8 + i] = hi[i]; }
    return r;
}

#if USE_TDM
// Issue a 2D TDM tile load (rows x rowlen bf16 elements) into LDS.
// D# layout per CDNA5 ISA 8.3-8.5: group0 = {flags, lds_addr, global_addr, type=2},
// group1 = {data_size=2B, tensor dims, tile dims, dim0 stride}.
static __device__ __forceinline__ void tdm_load_2d(unsigned lds_off, const void* gptr,
                                                   unsigned rowlen, unsigned rows,
                                                   unsigned long long stride_elems) {
    unsigned long long ga = (unsigned long long)(uintptr_t)gptr;
    u32x4 g0;
    g0.x = 1u;                                                  // count=1, user descriptor
    g0.y = lds_off;                                             // LDS byte address
    g0.z = (unsigned)ga;                                        // global_addr[31:0]
    g0.w = (unsigned)((ga >> 32) & 0x01FFFFFFu) | (2u << 30);   // addr[56:32] | type=2
    i32x8 g1;
    g1[0] = 0x10000;                                            // data_size=1 -> 2 bytes
    g1[1] = (int)((rowlen & 0xFFFFu) << 16);                    // tensor_dim0[15:0]
    g1[2] = (int)((rowlen >> 16) | ((rows & 0xFFFFu) << 16));   // td0 hi | td1 lo
    g1[3] = (int)((rows >> 16) | ((rowlen & 0xFFFFu) << 16));   // td1 hi | tile_dim0
    g1[4] = (int)(rows & 0xFFFFu);                              // tile_dim1, tile_dim2=0
    g1[5] = (int)(stride_elems & 0xFFFFFFFFull);                // dim0 stride lo
    g1[6] = (int)((stride_elems >> 32) & 0xFFFFull);            // dim0 stride hi
    g1[7] = 0;
    i32x4 z4 = {0, 0, 0, 0};
#if defined(__clang_major__) && (__clang_major__ >= 23)
    i32x8 z8 = {0, 0, 0, 0, 0, 0, 0, 0};
    __builtin_amdgcn_tensor_load_to_lds(g0, g1, z4, z4, z8, 0);
#else
    __builtin_amdgcn_tensor_load_to_lds(g0, g1, z4, z4, 0);
#endif
}
#endif

// Stage a (rows x rowlen) bf16 panel from global into LDS. Caller provides barriers.
static __device__ __forceinline__ void stage_panel(__bf16* lds, const __bf16* g,
                                                   int rows, int rowlen, size_t gstride) {
#if USE_TDM
    if (threadIdx.x < 32) {
        tdm_load_2d((unsigned)(uintptr_t)lds, (const void*)g,
                    (unsigned)rowlen, (unsigned)rows, (unsigned long long)gstride);
        __builtin_amdgcn_s_wait_tensorcnt(0);
    }
#else
    int total = rows * rowlen / 8;
    for (int i = (int)threadIdx.x; i < total; i += (int)blockDim.x) {
        int e = i * 8, r = e / rowlen, c = e - r * rowlen;
        *(v8bf*)(lds + (size_t)r * rowlen + c) = *(const v8bf*)(g + (size_t)r * gstride + c);
    }
#endif
}

// ---- prep: bf16 weights + padded bf16 input, bias sum, zero state ----------
__global__ void prep_kernel(const float* __restrict__ x, const float* __restrict__ Wd,
                            const float* __restrict__ bd, const float* __restrict__ Wf,
                            const float* __restrict__ Wr,
                            __bf16* __restrict__ xbf, __bf16* __restrict__ Wdbf,
                            __bf16* __restrict__ Wfbf, __bf16* __restrict__ Wrbf,
                            float* __restrict__ bdsum, float* __restrict__ m,
                            __bf16* __restrict__ sbf, float* __restrict__ obuf) {
    int tid = blockIdx.x * blockDim.x + threadIdx.x;
    int nth = gridDim.x * blockDim.x;
    for (long long i = tid; i < (long long)BATCH * WIN * KP; i += nth) {
        int k = (int)(i % KP);
        long long bt = i / KP;
        xbf[i] = (k < NUM_INPUT) ? f2bf(x[bt * NUM_INPUT + k]) : bf_zero();
    }
    for (int i = tid; i < 3 * H * KP; i += nth) {
        int k = i % KP, rh = i / KP;            // rh = d*H + h
        Wdbf[i] = (k < NUM_INPUT) ? f2bf(Wd[(size_t)rh * NUM_INPUT + k]) : bf_zero();
    }
    for (int i = tid; i < NLAYERS * H * H; i += nth) {
        Wfbf[i] = f2bf(Wf[i]);
        Wrbf[i] = f2bf(Wr[i]);
    }
    for (int i = tid; i < H; i += nth)
        bdsum[i] = bd[i] + bd[H + i] + bd[2 * H + i];
    for (int i = tid; i < NLAYERS * BATCH * H; i += nth) { m[i] = 0.f; sbf[i] = bf_zero(); }
    for (int i = tid; i < 4 * BATCH * NOUT; i += nth) obuf[i] = 0.f;
}

// ---- delayed-input GEMM: cur[t][b][h] (bf16, incl. summed delay biases) ----
__global__ void cur_kernel(const __bf16* __restrict__ xbf, const __bf16* __restrict__ Wdbf,
                           const float* __restrict__ bdsum, __bf16* __restrict__ curbf) {
    __shared__ __align__(16) __bf16 sA[16 * KP];   // one 16-row A panel, full K
    const int DLY[3] = {0, 16, 32};
    int b0 = blockIdx.x * 16;
    int t  = blockIdx.y;
    int wave = threadIdx.x >> 5, lane = threadIdx.x & 31;
    v8f acc[2] = {};
#pragma unroll
    for (int i = 0; i < 3; ++i) {
        int td = t - DLY[i];
        if (td < 0) continue;                   // zero-padded region contributes nothing
        __syncthreads();                        // previous panel fully consumed
        stage_panel(sA, xbf + ((size_t)b0 * WIN + td) * KP, 16, KP, (size_t)WIN * KP);
        __syncthreads();                        // panel visible to all waves
        const __bf16* wb = Wdbf + (size_t)i * H * KP;
#pragma unroll
        for (int kb = 0; kb < KP; kb += 32) {
            v16bf a = load_a_bf16(sA + kb, KP, lane);      // ds_load from LDS panel
#pragma unroll
            for (int ct = 0; ct < 2; ++ct) {
                int n0 = wave * 32 + ct * 16;
                const __bf16* bp = wb + (size_t)n0 * KP + kb;
                if (kb + 32 < KP) __builtin_prefetch(bp + 32, 0, 2);
                acc[ct] = wmma_bf16(a, load_b_bf16(bp, KP, lane), acc[ct]);
            }
        }
    }
    int n = lane & 15, mb = (lane >> 4) * 8;
#pragma unroll
    for (int ct = 0; ct < 2; ++ct) {
        int h = wave * 32 + ct * 16 + n;
        float bias = bdsum[h];
#pragma unroll
        for (int r = 0; r < 8; ++r) {
            int b = b0 + mb + r;
            curbf[((size_t)t * BATCH + b) * H + h] = f2bf(acc[ct][r] + bias);
        }
    }
}

// ---- one recurrent layer, one timestep -------------------------------------
__global__ void layer_kernel(const __bf16* __restrict__ prevbf,   // [BATCH][H] bf16
                             __bf16* __restrict__ sl,             // this layer's spikes
                             float* __restrict__ ml,              // this layer's membrane
                             const __bf16* __restrict__ Wf,
                             const __bf16* __restrict__ Wr,
                             const float* __restrict__ bias) {
    __shared__ __align__(16) __bf16 sP[16 * H];   // feedforward activations panel
    __shared__ __align__(16) __bf16 sS[16 * H];   // own-spike panel (snapshot)
    int b0 = blockIdx.x * 16;
    int wave = threadIdx.x >> 5, lane = threadIdx.x & 31;
    stage_panel(sP, prevbf + (size_t)b0 * H, 16, H, (size_t)H);
    stage_panel(sS, sl + (size_t)b0 * H, 16, H, (size_t)H);
    __syncthreads();
    v8f acc[2] = {};
#pragma unroll
    for (int kb = 0; kb < H; kb += 32) {
        v16bf aff = load_a_bf16(sP + kb, H, lane);
        v16bf arc = load_a_bf16(sS + kb, H, lane);
#pragma unroll
        for (int ct = 0; ct < 2; ++ct) {
            int n0 = wave * 32 + ct * 16;
            const __bf16* fp = Wf + (size_t)n0 * H + kb;
            const __bf16* rp = Wr + (size_t)n0 * H + kb;
            if (kb + 32 < H) {
                __builtin_prefetch(fp + 32, 0, 2);
                __builtin_prefetch(rp + 32, 0, 2);
            }
            acc[ct] = wmma_bf16(aff, load_b_bf16(fp, H, lane), acc[ct]);
            acc[ct] = wmma_bf16(arc, load_b_bf16(rp, H, lane), acc[ct]);
        }
    }
    // LDS snapshot decouples spike reads from the global writes below: no barrier.
    int n = lane & 15, mb = (lane >> 4) * 8;
#pragma unroll
    for (int ct = 0; ct < 2; ++ct) {
        int h = wave * 32 + ct * 16 + n;
        float bs = bias[h];
#pragma unroll
        for (int r = 0; r < 8; ++r) {
            int b = b0 + mb + r;
            size_t idx = (size_t)b * H + h;
            float mold = ml[idx];
            float sold = bf2f(sS[(size_t)(mb + r) * H + h]);   // pre-update spike
            float mnew = mold * ALPHA * (1.f - sold) + acc[ct][r] + bs;
            ml[idx] = mnew;
            sl[idx] = ((mnew - THRESH) > 0.f) ? f2bf(1.f) : bf_zero();
        }
    }
}

// ---- readout: membrane, spike, spike-count, softmax accumulation -----------
__global__ void readout_kernel(const __bf16* __restrict__ s1,
                               const float* __restrict__ Wo, const float* __restrict__ bo,
                               float* __restrict__ om, float* __restrict__ os,
                               float* __restrict__ osum, float* __restrict__ omot) {
    int b = blockIdx.x * blockDim.x + threadIdx.x;
    if (b >= BATCH) return;
    const __bf16* srow = s1 + (size_t)b * H;
    float mem[NOUT];
    for (int j = 0; j < NOUT; ++j) {
        float dot = bo[j];
        const float* w = Wo + (size_t)j * H;
#pragma unroll 8
        for (int h = 0; h < H; ++h) dot += bf2f(srow[h]) * w[h];
        size_t idx = (size_t)b * NOUT + j;
        float mv = om[idx] * ALPHA * (1.f - os[idx]) + dot;
        om[idx] = mv;
        float sp = ((mv - THRESH) > 0.f) ? 1.f : 0.f;
        os[idx] = sp;
        osum[idx] += sp;
        mem[j] = mv;
    }
    float mx = mem[0];
#pragma unroll
    for (int j = 1; j < NOUT; ++j) mx = fmaxf(mx, mem[j]);
    float se = 0.f, ex[NOUT];
#pragma unroll
    for (int j = 0; j < NOUT; ++j) { ex[j] = __expf(mem[j] - mx); se += ex[j]; }
    float inv = 1.f / se;
#pragma unroll
    for (int j = 0; j < NOUT; ++j) omot[(size_t)b * NOUT + j] += ex[j] * inv;
}

__global__ void finalize_kernel(const float* __restrict__ osum,
                                const float* __restrict__ omot, float* __restrict__ out) {
    int i = blockIdx.x * blockDim.x + threadIdx.x;
    if (i < BATCH * NOUT) {
        out[i] = osum[i] * (1.f / WIN);
        out[BATCH * NOUT + i] = omot[i];
    }
}

extern "C" void kernel_launch(void* const* d_in, const int* in_sizes, int n_in,
                              void* d_out, int out_size, void* d_ws, size_t ws_size,
                              hipStream_t stream) {
    const float* x   = (const float*)d_in[0];
    const float* Wd  = (const float*)d_in[1];
    const float* bd  = (const float*)d_in[2];
    const float* Wf  = (const float*)d_in[3];
    const float* bff = (const float*)d_in[4];
    const float* Wr  = (const float*)d_in[5];
    const float* Wo  = (const float*)d_in[6];
    const float* bo  = (const float*)d_in[7];

    char* ws = (char*)d_ws;
    size_t off = 0;
    auto alloc = [&](size_t bytes) {
        void* p = ws + off;
        off = (off + bytes + 255) & ~(size_t)255;
        return p;
    };
    __bf16* xbf   = (__bf16*)alloc((size_t)BATCH * WIN * KP * 2);
    __bf16* Wdbf  = (__bf16*)alloc((size_t)3 * H * KP * 2);
    __bf16* Wfbf  = (__bf16*)alloc((size_t)NLAYERS * H * H * 2);
    __bf16* Wrbf  = (__bf16*)alloc((size_t)NLAYERS * H * H * 2);
    float*  bdsum = (float*)alloc((size_t)H * 4);
    __bf16* curbf = (__bf16*)alloc((size_t)WIN * BATCH * H * 2);
    float*  mbuf  = (float*)alloc((size_t)NLAYERS * BATCH * H * 4);
    __bf16* sbf   = (__bf16*)alloc((size_t)NLAYERS * BATCH * H * 2);
    float*  obuf  = (float*)alloc((size_t)4 * BATCH * NOUT * 4);
    float*  om    = obuf;
    float*  osp   = obuf + 1 * BATCH * NOUT;
    float*  osum  = obuf + 2 * BATCH * NOUT;
    float*  omot  = obuf + 3 * BATCH * NOUT;

    prep_kernel<<<dim3(1024), dim3(256), 0, stream>>>(x, Wd, bd, Wf, Wr, xbf, Wdbf, Wfbf,
                                                      Wrbf, bdsum, mbuf, sbf, obuf);
    cur_kernel<<<dim3(BATCH / 16, WIN), dim3(256), 0, stream>>>(xbf, Wdbf, bdsum, curbf);

    for (int t = 0; t < WIN; ++t) {
        // layer 0: feedforward from delayed-input currents
        layer_kernel<<<dim3(BATCH / 16), dim3(256), 0, stream>>>(
            curbf + (size_t)t * BATCH * H, sbf, mbuf, Wfbf, Wrbf, bff);
        // layer 1: feedforward from layer-0 spikes (just updated)
        layer_kernel<<<dim3(BATCH / 16), dim3(256), 0, stream>>>(
            sbf, sbf + (size_t)BATCH * H, mbuf + (size_t)BATCH * H,
            Wfbf + (size_t)H * H, Wrbf + (size_t)H * H, bff + H);
        readout_kernel<<<dim3(2), dim3(256), 0, stream>>>(
            sbf + (size_t)BATCH * H, Wo, bo, om, osp, osum, omot);
    }
    finalize_kernel<<<dim3((BATCH * NOUT + 255) / 256), dim3(256), 0, stream>>>(
        osum, omot, (float*)d_out);
}
